// Model_25563645346643
// MI455X (gfx1250) — matrix operand order repaired and checked by reading, not compiled, not run on hardware
//
#include <hip/hip_runtime.h>

#define B_  32
#define L_  512
#define CH_ 512
#define NH_ 8
#define HD_ 64
#define NB_ 2
#define NEGV (-4294967295.0f)

typedef __attribute__((ext_vector_type(16))) __bf16 v16bf;
typedef __attribute__((ext_vector_type(8)))  float  v8f;

union Frag {
    v16bf v;
    uint4 u[2];
    unsigned short s[16];
};

struct FragSet {
    Frag a[2];
    Frag b[4];
};

__device__ __forceinline__ unsigned short f2bf(float x) {
    unsigned int u = __float_as_uint(x);
    unsigned int r = (u + 0x7FFFu + ((u >> 16) & 1u)) >> 16;
    return (unsigned short)r;
}
__device__ __forceinline__ float bf2f(unsigned short h) {
    return __uint_as_float(((unsigned int)h) << 16);
}
__device__ __forceinline__ v8f wmma_bf16(const Frag& a, const Frag& b, v8f c) {
    return __builtin_amdgcn_wmma_f32_16x16x32_bf16(false, a.v, false, b.v, (short)0, c, false, false);
}

// ---------------- f32 -> bf16 conversion ----------------
__global__ void cvt_kernel(const float* __restrict__ in, unsigned short* __restrict__ out, long n) {
    long i = (long)blockIdx.x * blockDim.x + threadIdx.x;
    long stride = (long)gridDim.x * blockDim.x;
    for (; i < n; i += stride) out[i] = f2bf(in[i]);
}

// ---------------- LayerNorm (optional residual add), C=512, block=256 ----------------
__global__ __launch_bounds__(256) void ln_kernel(
    const float* __restrict__ in, const float* __restrict__ addp,
    const float* __restrict__ g, const float* __restrict__ bta,
    float* __restrict__ outF, unsigned short* __restrict__ outB) {
    const int C = CH_;
    long row = blockIdx.x;
    int t = threadIdx.x;
    __shared__ float sh[256];
    const float* rin = in + row * (long)C;
    float x0 = rin[t], x1 = rin[t + 256];
    if (addp) {
        const float* ra = addp + row * (long)C;
        x0 += ra[t]; x1 += ra[t + 256];
    }
    sh[t] = x0 + x1;
    __syncthreads();
    for (int o = 128; o > 0; o >>= 1) { if (t < o) sh[t] += sh[t + o]; __syncthreads(); }
    float mean = sh[0] * (1.0f / C);
    __syncthreads();
    float d0 = x0 - mean, d1 = x1 - mean;
    sh[t] = d0 * d0 + d1 * d1;
    __syncthreads();
    for (int o = 128; o > 0; o >>= 1) { if (t < o) sh[t] += sh[t + o]; __syncthreads(); }
    float inv = rsqrtf(sh[0] * (1.0f / C) + 1e-8f);
    float y0 = d0 * inv * g[t] + bta[t];
    float y1 = d1 * inv * g[t + 256] + bta[t + 256];
    if (outF) { float* ro = outF + row * (long)C; ro[t] = y0; ro[t + 256] = y1; }
    if (outB) { unsigned short* ro = outB + row * (long)C; ro[t] = f2bf(y0); ro[t + 256] = f2bf(y1); }
}

// ---------------- bf16 WMMA GEMM: out[M,N] = A[M,K] @ W[N,K]^T ----------------
// Block tile 256x64 (8 waves), wave tile 32x64 (2 row-tiles x 4 col-chunks).
// Software-pipelined: k-tile t+1 fragments load while k-tile t WMMAs execute.
// Epilogue: +bias, +pos (bf16 [L_,N] at row%L_), relu/swish, +resid(f32),
//           stores f32 (outF), bf16 (outB), and/or bf16 transposed (outBT, [N, ldT]).
__global__ __launch_bounds__(256) void gemm_bf16_kernel(
    const unsigned short* __restrict__ A,
    const unsigned short* __restrict__ W, int ldw,
    const float* __restrict__ bias,
    const unsigned short* __restrict__ pos,
    const float* __restrict__ resid,
    float* __restrict__ outF, unsigned short* __restrict__ outB,
    unsigned short* __restrict__ outBT, int ldT,
    int M, int N, int K, int flags) {            // flags: 1=relu, 2=swish
    int lane = threadIdx.x & 31;
    int wave = threadIdx.x >> 5;
    int m0 = blockIdx.y * 256 + wave * 32;
    int n0 = blockIdx.x * 64;
    int ln = lane & 15;
    int half = lane >> 4;
    int kbA = half * 8;
    int kbB = half * 8;

    long r0 = m0 + ln;      if (r0 >= M) r0 = M - 1;
    long r1 = m0 + 16 + ln; if (r1 >= M) r1 = M - 1;
    const unsigned short* Arow0 = A + r0 * (long)K;
    const unsigned short* Arow1 = A + r1 * (long)K;
    const unsigned short* Wrow  = W + (long)(n0 + ln) * ldw + kbB;
    const long wstep = 16 * (long)ldw;

    v8f acc[2][4];
#pragma unroll
    for (int t = 0; t < 2; t++)
#pragma unroll
        for (int c = 0; c < 4; c++) acc[t][c] = {};

    auto loadf = [&](int kk, FragSet& f) {
        f.a[0].u[0] = *(const uint4*)(Arow0 + kk + kbA);
        f.a[0].u[1] = *(const uint4*)(Arow0 + kk + 16 + kbA);
        f.a[1].u[0] = *(const uint4*)(Arow1 + kk + kbA);
        f.a[1].u[1] = *(const uint4*)(Arow1 + kk + 16 + kbA);
#pragma unroll
        for (int c = 0; c < 4; c++) {
            const unsigned short* wb = Wrow + (long)c * wstep + kk;
            f.b[c].u[0] = *(const uint4*)(wb);
            f.b[c].u[1] = *(const uint4*)(wb + 16);
        }
    };
    auto dowmma = [&](FragSet& f) {
#pragma unroll
        for (int t = 0; t < 2; t++)
#pragma unroll
            for (int c = 0; c < 4; c++)
                acc[t][c] = wmma_bf16(f.a[t], f.b[c], acc[t][c]);
    };

    FragSet f0, f1;
    loadf(0, f0);
    int kk = 0;
    for (; kk + 64 < K; kk += 64) {
        loadf(kk + 32, f1);     // overlap with wmma(f0)
        dowmma(f0);
        loadf(kk + 64, f0);     // overlap with wmma(f1)
        dowmma(f1);
    }
    loadf(kk + 32, f1);
    dowmma(f0);
    dowmma(f1);

    int rbase = half * 8;
#pragma unroll
    for (int t = 0; t < 2; t++) {
#pragma unroll
        for (int c = 0; c < 4; c++) {
            int col = n0 + c * 16 + ln;
            float bv = bias ? bias[col] : 0.0f;
#pragma unroll
            for (int r = 0; r < 8; r++) {
                int row = m0 + t * 16 + rbase + r;
                if (row < M) {
                    float v = acc[t][c][r] + bv;
                    if (pos) v += bf2f(pos[(long)(row % L_) * N + col]);
                    if (flags & 1) v = fmaxf(v, 0.0f);
                    if (flags & 2) v = v / (1.0f + __expf(-v));   // swish
                    if (resid) v += resid[(long)row * N + col];
                    if (outF) outF[(long)row * N + col] = v;
                    if (outB) outB[(long)row * N + col] = f2bf(v);
                    if (outBT) outBT[(long)col * ldT + row] = f2bf(v);
                }
            }
        }
    }
}

// ---------------- Flash-style causal attention ----------------
// One wave = 16 query columns of one (b,h).  Computes S^T = K@Q^T and
// O^T = V^T @ P^T so every global access is a contiguous b128 load, softmax
// stats are per-lane scalars, and P feeds the B-fragment with no exchange.
// K already includes posK; VT is V+posV stored transposed as [CH_, B_*L_].
__global__ __launch_bounds__(128) void attn_kernel(
    const unsigned short* __restrict__ Q,
    const unsigned short* __restrict__ Kp,
    const unsigned short* __restrict__ VT,
    float* __restrict__ O) {
    const float scale = 0.125f;   // 1/sqrt(HD=64)
    const int NTOK = B_ * L_;
    int lane = threadIdx.x & 31;
    int wave = threadIdx.x >> 5;
    int gw = blockIdx.x * 4 + wave;       // B_*NH_*(L_/16) = 8192 waves
    int lt = gw & 31;
    int h  = (gw >> 5) & (NH_ - 1);
    int b  = gw >> 8;
    int l0 = lt * 16;
    int ln = lane & 15;
    int half = lane >> 4;
    int kbA = half * 8;
    int kbB = half * 8;
    int rbase = half * 8;

    long base = ((long)b * L_) * CH_ + (long)h * HD_;
    int myq = l0 + ln;                    // query column owned by this lane

    // Q as B-fragments (two 32-wide d chunks)
    Frag qb0, qb1;
    {
        const unsigned short* qr = Q + base + (long)myq * CH_ + kbB;
        qb0.u[0] = *(const uint4*)(qr);
        qb0.u[1] = *(const uint4*)(qr + 16);
        qb1.u[0] = *(const uint4*)(qr + 32);
        qb1.u[1] = *(const uint4*)(qr + 48);
    }

    v8f o0 = {}, o1 = {}, o2 = {}, o3 = {};
    float mrun = -3.0e38f, lrun = 0.0f;

    for (int m0 = 0; m0 <= l0 + 15; m0 += 32) {
        // ---- S^T = K_tile @ Q^T : s0 = keys m0..+15, s1 = keys m0+16..+31
        v8f s0 = {}, s1 = {};
        {
            const unsigned short* kr0 = Kp + base + (long)(m0 + ln) * CH_;
            const unsigned short* kr1 = kr0 + 16 * CH_;
            Frag ka;
            ka.u[0] = *(const uint4*)(kr0 + kbA);
            ka.u[1] = *(const uint4*)(kr0 + 16 + kbA);
            s0 = wmma_bf16(ka, qb0, s0);
            ka.u[0] = *(const uint4*)(kr0 + 32 + kbA);
            ka.u[1] = *(const uint4*)(kr0 + 48 + kbA);
            s0 = wmma_bf16(ka, qb1, s0);
            ka.u[0] = *(const uint4*)(kr1 + kbA);
            ka.u[1] = *(const uint4*)(kr1 + 16 + kbA);
            s1 = wmma_bf16(ka, qb0, s1);
            ka.u[0] = *(const uint4*)(kr1 + 32 + kbA);
            ka.u[1] = *(const uint4*)(kr1 + 48 + kbA);
            s1 = wmma_bf16(ka, qb1, s1);
        }

        // ---- V^T A-fragments: issue early, independent of softmax VALU work
        Frag va0, va1, va2, va3;
        {
            const unsigned short* vr = VT + (long)(h * HD_ + ln) * NTOK + (long)b * L_ + m0 + kbA;
            const long cstep = 16 * (long)NTOK;
            va0.u[0] = *(const uint4*)(vr);
            va0.u[1] = *(const uint4*)(vr + 16);
            va1.u[0] = *(const uint4*)(vr + cstep);
            va1.u[1] = *(const uint4*)(vr + cstep + 16);
            va2.u[0] = *(const uint4*)(vr + 2 * cstep);
            va2.u[1] = *(const uint4*)(vr + 2 * cstep + 16);
            va3.u[0] = *(const uint4*)(vr + 3 * cstep);
            va3.u[1] = *(const uint4*)(vr + 3 * cstep + 16);
        }

        // ---- online softmax: per-lane stats for query column `myq`
        float p0[8], p1[8];
        float mloc = -3.0e38f;
#pragma unroll
        for (int r = 0; r < 8; r++) {
            float a0 = s0[r] * scale; if (m0 + rbase + r > myq)      a0 = NEGV;
            float a1 = s1[r] * scale; if (m0 + 16 + rbase + r > myq) a1 = NEGV;
            p0[r] = a0; p1[r] = a1;
            mloc = fmaxf(mloc, fmaxf(a0, a1));
        }
        mloc = fmaxf(mloc, __shfl_xor(mloc, 16, 32));   // combine half-wave key ranges
        float mn = fmaxf(mrun, mloc);
        float alpha = __expf(mrun - mn);
        float sloc = 0.0f;
#pragma unroll
        for (int r = 0; r < 8; r++) {
            p0[r] = __expf(p0[r] - mn);
            p1[r] = __expf(p1[r] - mn);
            sloc += p0[r] + p1[r];
        }
        sloc += __shfl_xor(sloc, 16, 32);
        mrun = mn;
        lrun = lrun * alpha + sloc;
#pragma unroll
        for (int r = 0; r < 8; r++) {
            o0[r] *= alpha; o1[r] *= alpha; o2[r] *= alpha; o3[r] *= alpha;
        }

        // ---- build P^T B-fragment: no exchange, no LDS round-trip
        // half0 lane needs keys {0..7}=own p0, {16..23}=own p1
        // half1 lane needs keys {8..15}=own p0, {24..31}=own p1
        Frag pb;
#pragma unroll
        for (int r = 0; r < 8; r++) {
            pb.s[r]     = f2bf(p0[r]);
            pb.s[r + 8] = f2bf(p1[r]);
        }

        // ---- O^T += V^T_tile @ P^T
        o0 = wmma_bf16(va0, pb, o0);
        o1 = wmma_bf16(va1, pb, o1);
        o2 = wmma_bf16(va2, pb, o2);
        o3 = wmma_bf16(va3, pb, o3);
    }

    // ---- normalize and store: per lane 4 runs of 8 consecutive d-values of one token
    float inv = 1.0f / lrun;
    float* orow = O + base + (long)myq * CH_;
#pragma unroll
    for (int c = 0; c < 4; c++) {
        const v8f& oc = (c == 0 ? o0 : c == 1 ? o1 : c == 2 ? o2 : o3);
        float4 t0, t1;
        t0.x = oc[0] * inv; t0.y = oc[1] * inv; t0.z = oc[2] * inv; t0.w = oc[3] * inv;
        t1.x = oc[4] * inv; t1.y = oc[5] * inv; t1.z = oc[6] * inv; t1.w = oc[7] * inv;
        *(float4*)(orow + c * 16 + rbase)     = t0;
        *(float4*)(orow + c * 16 + rbase + 4) = t1;
    }
}

// ---------------- diffusion timestep sin/cos embedding -> bf16 [32, 512] ----------------
__global__ void temb_embed_kernel(const int* __restrict__ t, unsigned short* __restrict__ out) {
    int b = blockIdx.x;
    int j = threadIdx.x;  // 256 = half
    float tv = (float)t[b];
    float fr = __expf((float)j * (-9.21034037197618f / 255.0f));
    float ang = tv * fr;
    out[b * 512 + j]       = f2bf(sinf(ang));
    out[b * 512 + 256 + j] = f2bf(cosf(ang));
}

// ---------------- x_t = seqs_nxt + temb[b] (bf16 out) ----------------
__global__ void xt_kernel(const float* __restrict__ nxt, const float* __restrict__ temb,
                          unsigned short* __restrict__ out) {
    long n = (long)B_ * L_ * CH_;
    long i = (long)blockIdx.x * blockDim.x + threadIdx.x;
    long stride = (long)gridDim.x * blockDim.x;
    for (; i < n; i += stride) {
        int c = (int)(i & (CH_ - 1));
        int b = (int)(i / ((long)L_ * CH_));
        out[i] = f2bf(nxt[i] + temb[b * CH_ + c]);
    }
}

extern "C" void kernel_launch(void* const* d_in, const int* in_sizes, int n_in,
                              void* d_out, int out_size, void* d_ws, size_t ws_size,
                              hipStream_t stream) {
    const float* log_seqs = (const float*)d_in[0];
    const float* seqs_nxt = (const float*)d_in[1];
    const int*   tvec     = (const int*)d_in[2];
    const float* qw  = (const float*)d_in[3];  const float* qb  = (const float*)d_in[4];
    const float* kw  = (const float*)d_in[5];  const float* kb  = (const float*)d_in[6];
    const float* vw  = (const float*)d_in[7];  const float* vb  = (const float*)d_in[8];
    const float* ln1g = (const float*)d_in[9];  const float* ln1b = (const float*)d_in[10];
    const float* ln2g = (const float*)d_in[11]; const float* ln2b = (const float*)d_in[12];
    const float* c1w = (const float*)d_in[13]; const float* c1b = (const float*)d_in[14];
    const float* c2w = (const float*)d_in[15]; const float* c2b = (const float*)d_in[16];
    const float* posK = (const float*)d_in[17]; const float* posV = (const float*)d_in[18];
    const float* lng = (const float*)d_in[19]; const float* lnb = (const float*)d_in[20];
    const float* tw0 = (const float*)d_in[21]; const float* tb0 = (const float*)d_in[22];
    const float* tw1 = (const float*)d_in[23]; const float* tb1 = (const float*)d_in[24];
    const float* tw2 = (const float*)d_in[25]; const float* tb2 = (const float*)d_in[26];
    const float* decw = (const float*)d_in[27]; const float* decb = (const float*)d_in[28];

    const long NTOK = (long)B_ * L_;     // 16384
    const long NEL  = NTOK * CH_;        // 8388608

    char* ws = (char*)d_ws;
    size_t off = 0;
    auto alloc = [&](size_t bytes) -> char* {
        char* p = ws + off;
        off += (bytes + 255) & ~(size_t)255;
        return p;
    };

    float* seqs_f = (float*)alloc(NEL * 4);
    float* qin_f  = (float*)alloc(NEL * 4);
    float* attn_f = (float*)alloc(NEL * 4);   // also reused as decoder accumulator
    unsigned short* seqs_bf = (unsigned short*)alloc(NEL * 2);
    unsigned short* a_bf    = (unsigned short*)alloc(NEL * 2);
    unsigned short* q_bf    = (unsigned short*)alloc(NEL * 2);
    unsigned short* k_bf    = (unsigned short*)alloc(NEL * 2);
    unsigned short* v_bf    = (unsigned short*)alloc(NEL * 2);  // holds V^T [CH_, NTOK]
    unsigned short* h_bf  = q_bf;   // reuse: q consumed by attention before FFN
    unsigned short* lf_bf = k_bf;   // reuse: k consumed before final LN
    unsigned short* xt_bf = v_bf;   // reuse: v consumed before decoder

    const long WQ = (long)NB_ * CH_ * CH_;
    unsigned short* qw_bf  = (unsigned short*)alloc(WQ * 2);
    unsigned short* kw_bf  = (unsigned short*)alloc(WQ * 2);
    unsigned short* vw_bf  = (unsigned short*)alloc(WQ * 2);
    unsigned short* c1w_bf = (unsigned short*)alloc(WQ * 2);
    unsigned short* c2w_bf = (unsigned short*)alloc(WQ * 2);
    unsigned short* posK_bf = (unsigned short*)alloc((long)L_ * CH_ * 2);
    unsigned short* posV_bf = (unsigned short*)alloc((long)L_ * CH_ * 2);
    unsigned short* tw0_bf = (unsigned short*)alloc((long)2048 * 512 * 2);
    unsigned short* tw1_bf = (unsigned short*)alloc((long)2048 * 2048 * 2);
    unsigned short* tw2_bf = (unsigned short*)alloc((long)512 * 2048 * 2);
    unsigned short* decw_bf = (unsigned short*)alloc((long)512 * 1024 * 2);
    unsigned short* te0 = (unsigned short*)alloc((long)32 * 512 * 2);
    unsigned short* te1 = (unsigned short*)alloc((long)32 * 2048 * 2);
    unsigned short* te2 = (unsigned short*)alloc((long)32 * 2048 * 2);
    float* temb_f = (float*)alloc((long)32 * 512 * 4);

    auto cvt = [&](const float* in, unsigned short* outp, long n) {
        long blocks = (n + 256 * 8 - 1) / (256 * 8);
        if (blocks > 2048) blocks = 2048;
        if (blocks < 1) blocks = 1;
        cvt_kernel<<<(int)blocks, 256, 0, stream>>>(in, outp, n);
    };
    cvt(log_seqs, seqs_bf, NEL);
    hipMemcpyAsync(seqs_f, log_seqs, NEL * 4, hipMemcpyDeviceToDevice, stream);
    cvt(qw, qw_bf, WQ); cvt(kw, kw_bf, WQ); cvt(vw, vw_bf, WQ);
    cvt(c1w, c1w_bf, WQ); cvt(c2w, c2w_bf, WQ);
    cvt(posK, posK_bf, (long)L_ * CH_); cvt(posV, posV_bf, (long)L_ * CH_);
    cvt(tw0, tw0_bf, (long)2048 * 512); cvt(tw1, tw1_bf, (long)2048 * 2048);
    cvt(tw2, tw2_bf, (long)512 * 2048); cvt(decw, decw_bf, (long)512 * 1024);

    auto gemm = [&](const unsigned short* A, const unsigned short* W, int ldw,
                    const float* bias, const unsigned short* pos, const float* resid,
                    float* outF, unsigned short* outB, unsigned short* outBT, int ldT,
                    int M, int N, int K, int flags) {
        dim3 grid(N / 64, (M + 255) / 256);
        gemm_bf16_kernel<<<grid, 256, 0, stream>>>(A, W, ldw, bias, pos, resid,
                                                   outF, outB, outBT, ldT, M, N, K, flags);
    };

    for (int i = 0; i < NB_; i++) {
        long wo = (long)i * CH_ * CH_;
        // Qin = LN1(seqs); keep f32 (residual) + bf16 (GEMM input)
        ln_kernel<<<(int)NTOK, 256, 0, stream>>>(seqs_f, nullptr, ln1g + i * CH_, ln1b + i * CH_, qin_f, a_bf);
        // Q = Qin@qw^T+qb ; K = seqs@kw^T+kb+posK ; V^T = (seqs@vw^T+vb+posV)^T
        gemm(a_bf,    qw_bf + wo, CH_, qb + i * CH_, nullptr, nullptr,
             nullptr, q_bf, nullptr, 0, (int)NTOK, CH_, CH_, 0);
        gemm(seqs_bf, kw_bf + wo, CH_, kb + i * CH_, posK_bf, nullptr,
             nullptr, k_bf, nullptr, 0, (int)NTOK, CH_, CH_, 0);
        gemm(seqs_bf, vw_bf + wo, CH_, vb + i * CH_, posV_bf, nullptr,
             nullptr, nullptr, v_bf, (int)NTOK, (int)NTOK, CH_, CH_, 0);
        attn_kernel<<<2048, 128, 0, stream>>>(q_bf, k_bf, v_bf, attn_f);
        // seqs = LN2(Qin + attn_out)
        ln_kernel<<<(int)NTOK, 256, 0, stream>>>(attn_f, qin_f, ln2g + i * CH_, ln2b + i * CH_, seqs_f, seqs_bf);
        // FFN with inner residual
        gemm(seqs_bf, c1w_bf + wo, CH_, c1b + i * CH_, nullptr, nullptr,
             nullptr, h_bf, nullptr, 0, (int)NTOK, CH_, CH_, 1);
        gemm(h_bf,    c2w_bf + wo, CH_, c2b + i * CH_, nullptr, seqs_f,
             seqs_f, seqs_bf, nullptr, 0, (int)NTOK, CH_, CH_, 0);
    }

    float* log_feats_out = (float*)d_out;
    float* dec_out = (float*)d_out + NEL;
    ln_kernel<<<(int)NTOK, 256, 0, stream>>>(seqs_f, nullptr, lng, lnb, log_feats_out, lf_bf);

    // timestep MLP (tiny): sin/cos embed -> 3 GEMMs with swish
    temb_embed_kernel<<<32, 256, 0, stream>>>(tvec, te0);
    gemm(te0, tw0_bf, 512,  tb0, nullptr, nullptr, nullptr, te1, nullptr, 0, 32, 2048, 512, 2);
    gemm(te1, tw1_bf, 2048, tb1, nullptr, nullptr, nullptr, te2, nullptr, 0, 32, 2048, 2048, 2);
    gemm(te2, tw2_bf, 2048, tb2, nullptr, nullptr, temb_f, nullptr, nullptr, 0, 32, 512, 2048, 0);
    xt_kernel<<<2048, 256, 0, stream>>>(seqs_nxt, temb_f, xt_bf);

    // dec = [log_feats, x_t] @ decw^T + decb  (two K=512 GEMMs, second accumulates)
    float* dec_tmp = attn_f;
    gemm(lf_bf, decw_bf,       1024, decb,    nullptr, nullptr,
         dec_tmp, nullptr, nullptr, 0, (int)NTOK, CH_, CH_, 0);
    gemm(xt_bf, decw_bf + 512, 1024, nullptr, nullptr, dec_tmp,
         dec_out, nullptr, nullptr, 0, (int)NTOK, CH_, CH_, 0);
}
